// G_OracleAD_54589034332595
// MI455X (gfx1250) — compile-verified, matrix operand order checked by reading, not run on hardware
//
#include <hip/hip_runtime.h>
#include <hip/hip_bf16.h>

// ---------------- problem constants ----------------
#define Bc   32
#define Nc   64
#define Lc   128
#define Hc   128
#define G4c  512            // 4*H
#define BNc  2048           // B*N
#define MT   16             // batch rows per workgroup (LSTM kernels)

typedef __bf16 bf16_t;
typedef __attribute__((ext_vector_type(16))) __bf16 v16bf;
typedef __attribute__((ext_vector_type(8)))  __bf16 v8bf;
typedef __attribute__((ext_vector_type(8)))  float  v8f;

#define V8F_ZERO {0.f,0.f,0.f,0.f,0.f,0.f,0.f,0.f}

// ---- WMMA helpers (CDNA5 gfx1250, wave32) ----
__device__ __forceinline__ v8f wmma_bf16(v16bf a, v16bf b, v8f c) {
  // v_wmma_f32_16x16x32_bf16: (neg_a, A, neg_b, B, c_mod, C, reuse_a, reuse_b)
  return __builtin_amdgcn_wmma_f32_16x16x32_bf16(false, a, false, b, (short)0, c, false, false);
}

// Load a 16x32 bf16 fragment (A layout; B uses the mirrored layout with
// rows == output columns, i.e. weights stored [n][k] row-major).
// 16-bit A 16x32 VGPR map: lane<16 -> row=lane, K = kb..kb+7 (v0..3), kb+16..kb+23 (v4..7)
//                          lane>=16 -> row=lane-16, K = kb+8..kb+15, kb+24..kb+31
__device__ __forceinline__ v16bf load_frag(const bf16_t* base, int stride, int kbase) {
  const int lane = threadIdx.x & 31;
  const int row  = lane & 15;
  const int koff = (lane >> 4) << 3;
  const bf16_t* p = base + row * stride + kbase + koff;
  union { v8bf h[2]; v16bf f; } u;
  u.h[0] = *reinterpret_cast<const v8bf*>(p);        // 16B -> ds_load_b128
  u.h[1] = *reinterpret_cast<const v8bf*>(p + 16);   // +16 elems = +32B
  return u.f;
}

// Store a 16x16 f32 C/D tile: VGPR r: lane<16 -> (M=r, N=lane); lane>=16 -> (M=r+8, N=lane-16)
__device__ __forceinline__ void store_tile(float* dst, int stride, v8f d) {
  const int lane = threadIdx.x & 31;
  const int n  = lane & 15;
  const int mb = (lane >> 4) << 3;
#pragma unroll
  for (int r = 0; r < 8; ++r) dst[(mb + r) * stride + n] = d[r];
}

__device__ __forceinline__ float fsig(float x)  { return 1.0f / (1.0f + __expf(-x)); }
__device__ __forceinline__ float ftanh(float x) {
  float e = __expf(-2.0f * x);
  return (1.0f - e) / (1.0f + e);
}

// ---------------- fp32 -> bf16 weight convert ----------------
__global__ void __launch_bounds__(256) cvt_bf16_kernel(const float* __restrict__ src,
                                                       bf16_t* __restrict__ dst, int n) {
  int i = blockIdx.x * 256 + threadIdx.x;
  if (i < n) dst[i] = (bf16_t)src[i];
}

// =================================================================
// Encoder LSTM (input size 1) + fused online-softmax attention pooling.
// One workgroup = 16 batch rows, 256 threads (8 wave32).
// W_hh (bf16, [512][128]) resident in LDS for the whole time loop.
// =================================================================
#define ENC_LDS 198464
__global__ void __launch_bounds__(256) enc_lstm_kernel(
    const float* __restrict__ x,       // [BN, L]
    const float* __restrict__ wih,     // [512]
    const float* __restrict__ bih,
    const float* __restrict__ bhh,
    const float* __restrict__ poolw,   // [128]
    const float* __restrict__ poolb,   // [1]
    const bf16_t* __restrict__ whh_bf, // [512][128]
    float* __restrict__ c_out)         // [BN, 128]
{
  extern __shared__ char smem[];
  bf16_t* sWhh = (bf16_t*)smem;                          // 131072 B
  float*  sG   = (float*)(smem + 131072);                // [16][512] 32768 B
  float*  sHf  = (float*)(smem + 163840);                // [16][128]
  float*  sC   = sHf + MT * Hc;
  float*  sS   = sC  + MT * Hc;                          // pooled accum
  bf16_t* sHbf = (bf16_t*)(sS + MT * Hc);                // [16][128] bf16
  float*  sWih = (float*)((char*)sHbf + MT * Hc * 2);    // [512]
  float*  sGb  = sWih + G4c;                             // [512]
  float*  sPw  = sGb + G4c;                              // [128]
  float*  sXt  = sPw + Hc;                               // [16]
  float*  sRed = sXt + MT;                               // [256]
  float*  sM   = sRed + 256;                             // [16] running max
  float*  sD   = sM + MT;                                // [16] running denom
  float*  sSc  = sD + MT;                                // [16] scale
  float*  sE   = sSc + MT;                               // [16] exp(new)

  const int tid = threadIdx.x;
  const int wv  = tid >> 5;
  const int bn0 = blockIdx.x * MT;

  // stage weights/state into LDS
  for (int i = tid; i < (G4c * Hc) / 8; i += 256)
    ((v8bf*)sWhh)[i] = ((const v8bf*)whh_bf)[i];
  for (int i = tid; i < G4c; i += 256) { sWih[i] = wih[i]; sGb[i] = bih[i] + bhh[i]; }
  for (int i = tid; i < Hc; i += 256) sPw[i] = poolw[i];
  for (int i = tid; i < MT * Hc; i += 256) {
    sHf[i] = 0.f; sC[i] = 0.f; sS[i] = 0.f; sHbf[i] = (bf16_t)0.f;
  }
  if (tid < MT) { sM[tid] = -3.0e38f; sD[tid] = 0.f; }
  __syncthreads();

  const float pb = poolb[0];

  for (int t = 0; t < Lc; ++t) {
    if (tid < MT) {
      sXt[tid] = x[(size_t)(bn0 + tid) * Lc + t];
      if (t + 16 < Lc) __builtin_prefetch(&x[(size_t)(bn0 + tid) * Lc + t + 16], 0, 0);
    }
    // ---- g = h_bf16 @ W_hh^T  (wave wv owns gate columns [wv*64, wv*64+64)) ----
    v8f a0 = V8F_ZERO, a1 = V8F_ZERO, a2 = V8F_ZERO, a3 = V8F_ZERO;
#pragma unroll
    for (int kb = 0; kb < Hc; kb += 32) {
      v16bf af = load_frag(sHbf, Hc, kb);
      a0 = wmma_bf16(af, load_frag(sWhh + (wv * 64 +  0) * Hc, Hc, kb), a0);
      a1 = wmma_bf16(af, load_frag(sWhh + (wv * 64 + 16) * Hc, Hc, kb), a1);
      a2 = wmma_bf16(af, load_frag(sWhh + (wv * 64 + 32) * Hc, Hc, kb), a2);
      a3 = wmma_bf16(af, load_frag(sWhh + (wv * 64 + 48) * Hc, Hc, kb), a3);
    }
    store_tile(sG + wv * 64 +  0, G4c, a0);
    store_tile(sG + wv * 64 + 16, G4c, a1);
    store_tile(sG + wv * 64 + 32, G4c, a2);
    store_tile(sG + wv * 64 + 48, G4c, a3);
    __syncthreads();

    // ---- gate nonlinearity + state update (rank-1 input proj fused) ----
#pragma unroll
    for (int it = 0; it < 8; ++it) {
      int idx = tid + it * 256;
      int m = idx >> 7, k = idx & 127;
      float xi = sXt[m];
      float gi = sG[m * G4c +       k] + xi * sWih[      k] + sGb[      k];
      float gf = sG[m * G4c + 128 + k] + xi * sWih[128 + k] + sGb[128 + k];
      float gg = sG[m * G4c + 256 + k] + xi * sWih[256 + k] + sGb[256 + k];
      float go = sG[m * G4c + 384 + k] + xi * sWih[384 + k] + sGb[384 + k];
      float cc = fsig(gf) * sC[idx] + fsig(gi) * ftanh(gg);
      sC[idx] = cc;
      float hh = fsig(go) * ftanh(cc);
      sHf[idx] = hh; sHbf[idx] = (bf16_t)hh;
    }
    __syncthreads();

    // ---- pooling logit: dot(h_m, pool_w), 16 threads/row ----
    {
      int m = tid >> 4, sub = tid & 15;
      float p = 0.f;
#pragma unroll
      for (int u = 0; u < 8; ++u) p += sHf[m * Hc + sub * 8 + u] * sPw[sub * 8 + u];
      sRed[tid] = p;
    }
    __syncthreads();
    if (tid < MT) {
      float sc = pb;
#pragma unroll
      for (int i = 0; i < 16; ++i) sc += sRed[tid * 16 + i];
      float mo = sM[tid], nm = fmaxf(mo, sc);
      float scl = __expf(mo - nm), e = __expf(sc - nm);
      sM[tid] = nm; sD[tid] = sD[tid] * scl + e;
      sSc[tid] = scl; sE[tid] = e;
    }
    __syncthreads();
    // ---- online-softmax weighted accumulation ----
#pragma unroll
    for (int it = 0; it < 8; ++it) {
      int idx = tid + it * 256;
      int m = idx >> 7;
      sS[idx] = sS[idx] * sSc[m] + sE[m] * sHf[idx];
    }
    __syncthreads();
  }

  // pooled context c = sum_t softmax * h
  for (int it = 0; it < 8; ++it) {
    int idx = tid + it * 256;
    int m = idx >> 7, k = idx & 127;
    c_out[(size_t)(bn0 + m) * Hc + k] = sS[idx] / sD[m];
  }
}

// =================================================================
// Spatial self-attention over N=64 variables. One workgroup per batch b.
// =================================================================
#define ATT_LDS 189952
__global__ void __launch_bounds__(256) attn_kernel(
    const float* __restrict__ c_in,     // [BN,128]
    const bf16_t* __restrict__ wq_bf, const float* __restrict__ bq,
    const bf16_t* __restrict__ wk_bf, const float* __restrict__ bk,
    const bf16_t* __restrict__ wv_bf, const float* __restrict__ bv,
    float* __restrict__ cstar_out,      // [BN,128]  (d_out section)
    float* __restrict__ A_out)          // [B,64,64] (d_out section)
{
  extern __shared__ char smem[];
  bf16_t* sCb = (bf16_t*)smem;                      // [64][128] bf16
  float*  sQf = (float*)(smem + 16384);             // [64][128]
  float*  sKf = sQf + Nc * Hc;
  float*  sVf = sKf + Nc * Hc;
  bf16_t* sQb = (bf16_t*)(sVf + Nc * Hc);           // [64][128] bf16
  bf16_t* sKb = sQb + Nc * Hc;
  bf16_t* sVt = sKb + Nc * Hc;                      // [128][64] bf16 (V^T)
  float*  sS  = (float*)(sVt + Hc * Nc);            // [64][64]
  bf16_t* sAb = (bf16_t*)(sS + Nc * Nc);            // [64][64] bf16
  float*  sBq = (float*)(sAb + Nc * Nc);            // [128]
  float*  sBk = sBq + Hc;
  float*  sBv = sBk + Hc;

  const int tid = threadIdx.x;
  const int wv  = tid >> 5;
  const int b   = blockIdx.x;
  const int bn0 = b * Nc;

  for (int i = tid; i < Nc * Hc; i += 256) sCb[i] = (bf16_t)c_in[(size_t)bn0 * Hc + i];
  if (tid < Hc) { sBq[tid] = bq[tid]; sBk[tid] = bk[tid]; sBv[tid] = bv[tid]; }
  __syncthreads();

  // ---- Q,K,V projections: wave wv owns output-column tile wv ----
  for (int p = 0; p < 3; ++p) {
    const bf16_t* W = (p == 0) ? wq_bf : (p == 1) ? wk_bf : wv_bf;
    float* O = (p == 0) ? sQf : (p == 1) ? sKf : sVf;
#pragma unroll
    for (int mt = 0; mt < 4; ++mt) {
      v8f acc = V8F_ZERO;
#pragma unroll
      for (int kb = 0; kb < Hc; kb += 32) {
        v16bf af = load_frag(sCb + mt * 16 * Hc, Hc, kb);
        acc = wmma_bf16(af, load_frag(W + (wv * 16) * Hc, Hc, kb), acc);
      }
      store_tile(O + mt * 16 * Hc + wv * 16, Hc, acc);
    }
  }
  __syncthreads();

  // bias add + bf16 convert (V transposed for the A@V matmul)
  for (int i = tid; i < Nc * Hc; i += 256) {
    int n = i >> 7, d = i & 127;
    sQb[i] = (bf16_t)(sQf[i] + sBq[d]);
    sKb[i] = (bf16_t)(sKf[i] + sBk[d]);
    sVt[d * Nc + n] = (bf16_t)(sVf[i] + sBv[d]);
  }
  __syncthreads();

  // ---- scores = Q @ K^T : 16 tiles, 2 per wave ----
  {
    int mt = wv >> 1, ntb = (wv & 1) * 2;
    v8f s0 = V8F_ZERO, s1 = V8F_ZERO;
#pragma unroll
    for (int kb = 0; kb < Hc; kb += 32) {
      v16bf af = load_frag(sQb + mt * 16 * Hc, Hc, kb);
      s0 = wmma_bf16(af, load_frag(sKb + (ntb    ) * 16 * Hc, Hc, kb), s0);
      s1 = wmma_bf16(af, load_frag(sKb + (ntb + 1) * 16 * Hc, Hc, kb), s1);
    }
    store_tile(sS + mt * 16 * Nc + ntb * 16,       Nc, s0);
    store_tile(sS + mt * 16 * Nc + ntb * 16 + 16,  Nc, s1);
  }
  __syncthreads();

  // ---- row softmax (64 rows, one thread each), scale 1/sqrt(H) ----
  if (tid < Nc) {
    const float inv = 0.08838834764831845f;  // 1/sqrt(128)
    float mx = -3.0e38f;
    for (int j = 0; j < Nc; ++j) mx = fmaxf(mx, sS[tid * Nc + j] * inv);
    float sum = 0.f;
    for (int j = 0; j < Nc; ++j) {
      float e = __expf(sS[tid * Nc + j] * inv - mx);
      sS[tid * Nc + j] = e; sum += e;
    }
    float r = 1.0f / sum;
    for (int j = 0; j < Nc; ++j) {
      float aij = sS[tid * Nc + j] * r;
      sAb[tid * Nc + j] = (bf16_t)aij;
      A_out[(size_t)b * (Nc * Nc) + tid * Nc + j] = aij;
    }
  }
  __syncthreads();

  // ---- c_star = A @ V : wave wv owns d-column tile wv, K = 64 ----
#pragma unroll
  for (int mt = 0; mt < 4; ++mt) {
    v8f acc = V8F_ZERO;
#pragma unroll
    for (int kb = 0; kb < Nc; kb += 32) {
      v16bf af = load_frag(sAb + mt * 16 * Nc, Nc, kb);
      acc = wmma_bf16(af, load_frag(sVt + (wv * 16) * Nc, Nc, kb), acc);
    }
    store_tile(sQf + mt * 16 * Hc + wv * 16, Hc, acc);  // reuse sQf as staging
  }
  __syncthreads();
  for (int i = tid; i < Nc * Hc; i += 256)
    cstar_out[(size_t)bn0 * Hc + i] = sQf[i];
}

// =================================================================
// Decoder LSTM: constant per-step input projection computed once via WMMA,
// then persistent recurrence; recon/pred readouts fused.
// =================================================================
#define DEC_LDS 225280
__global__ void __launch_bounds__(256) dec_lstm_kernel(
    const float* __restrict__ cstar,    // [BN,128]
    const bf16_t* __restrict__ wih_bf,  // [512][128]
    const bf16_t* __restrict__ whh_bf,  // [512][128]
    const float* __restrict__ bih, const float* __restrict__ bhh,
    const float* __restrict__ recw, const float* __restrict__ recb,
    const float* __restrict__ predw, const float* __restrict__ predb,
    float* __restrict__ recon,          // [BN, L]
    float* __restrict__ pred)           // [BN]
{
  extern __shared__ char smem[];
  bf16_t* sWhh = (bf16_t*)smem;                       // 131072 B
  float*  sDin = (float*)(smem + 131072);             // [16][512]
  float*  sG   = sDin + MT * G4c;                     // [16][512]
  float*  sHf  = sG + MT * G4c;                       // [16][128]
  float*  sC   = sHf + MT * Hc;
  bf16_t* sHbf = (bf16_t*)(sC + MT * Hc);             // [16][128] bf16
  bf16_t* sCsb = sHbf + MT * Hc;                      // [16][128] bf16
  float*  sGb  = (float*)(sCsb + MT * Hc);            // [512]
  float*  sRw  = sGb + G4c;                           // [128]
  float*  sPw  = sRw + Hc;                            // [128]
  float*  sRed = sPw + Hc;                            // [256]

  const int tid = threadIdx.x;
  const int wv  = tid >> 5;
  const int bn0 = blockIdx.x * MT;

  for (int i = tid; i < MT * Hc; i += 256) {
    sCsb[i] = (bf16_t)cstar[(size_t)bn0 * Hc + i];
    sHf[i] = 0.f; sC[i] = 0.f; sHbf[i] = (bf16_t)0.f;
  }
  for (int i = tid; i < G4c; i += 256) sGb[i] = bih[i] + bhh[i];
  if (tid < Hc) { sRw[tid] = recw[tid]; sPw[tid] = predw[tid]; }
  __syncthreads();

  // ---- dec_in = c_star @ W_ih^T (one-time), W_ih fragments from global ----
  {
    v8f a0 = V8F_ZERO, a1 = V8F_ZERO, a2 = V8F_ZERO, a3 = V8F_ZERO;
#pragma unroll
    for (int kb = 0; kb < Hc; kb += 32) {
      v16bf af = load_frag(sCsb, Hc, kb);
      a0 = wmma_bf16(af, load_frag(wih_bf + (wv * 64 +  0) * Hc, Hc, kb), a0);
      a1 = wmma_bf16(af, load_frag(wih_bf + (wv * 64 + 16) * Hc, Hc, kb), a1);
      a2 = wmma_bf16(af, load_frag(wih_bf + (wv * 64 + 32) * Hc, Hc, kb), a2);
      a3 = wmma_bf16(af, load_frag(wih_bf + (wv * 64 + 48) * Hc, Hc, kb), a3);
    }
    store_tile(sDin + wv * 64 +  0, G4c, a0);
    store_tile(sDin + wv * 64 + 16, G4c, a1);
    store_tile(sDin + wv * 64 + 32, G4c, a2);
    store_tile(sDin + wv * 64 + 48, G4c, a3);
  }
  // stage W_hh into LDS (no conflict with the WMMA above)
  for (int i = tid; i < (G4c * Hc) / 8; i += 256)
    ((v8bf*)sWhh)[i] = ((const v8bf*)whh_bf)[i];
  __syncthreads();
  for (int i = tid; i < MT * G4c; i += 256) sDin[i] += sGb[i & 511];
  __syncthreads();

  const float rb = recb[0], pbv = predb[0];

  for (int t = 0; t < Lc; ++t) {
    // ---- g = h @ W_hh^T ----
    v8f a0 = V8F_ZERO, a1 = V8F_ZERO, a2 = V8F_ZERO, a3 = V8F_ZERO;
#pragma unroll
    for (int kb = 0; kb < Hc; kb += 32) {
      v16bf af = load_frag(sHbf, Hc, kb);
      a0 = wmma_bf16(af, load_frag(sWhh + (wv * 64 +  0) * Hc, Hc, kb), a0);
      a1 = wmma_bf16(af, load_frag(sWhh + (wv * 64 + 16) * Hc, Hc, kb), a1);
      a2 = wmma_bf16(af, load_frag(sWhh + (wv * 64 + 32) * Hc, Hc, kb), a2);
      a3 = wmma_bf16(af, load_frag(sWhh + (wv * 64 + 48) * Hc, Hc, kb), a3);
    }
    store_tile(sG + wv * 64 +  0, G4c, a0);
    store_tile(sG + wv * 64 + 16, G4c, a1);
    store_tile(sG + wv * 64 + 32, G4c, a2);
    store_tile(sG + wv * 64 + 48, G4c, a3);
    __syncthreads();

#pragma unroll
    for (int it = 0; it < 8; ++it) {
      int idx = tid + it * 256;
      int m = idx >> 7, k = idx & 127;
      float gi = sG[m * G4c +       k] + sDin[m * G4c +       k];
      float gf = sG[m * G4c + 128 + k] + sDin[m * G4c + 128 + k];
      float gg = sG[m * G4c + 256 + k] + sDin[m * G4c + 256 + k];
      float go = sG[m * G4c + 384 + k] + sDin[m * G4c + 384 + k];
      float cc = fsig(gf) * sC[idx] + fsig(gi) * ftanh(gg);
      sC[idx] = cc;
      float hh = fsig(go) * ftanh(cc);
      sHf[idx] = hh; sHbf[idx] = (bf16_t)hh;
    }
    __syncthreads();

    // ---- recon[bn,t] = dot(h, rec_w) + rec_b ----
    {
      int m = tid >> 4, sub = tid & 15;
      float p = 0.f;
#pragma unroll
      for (int u = 0; u < 8; ++u) p += sHf[m * Hc + sub * 8 + u] * sRw[sub * 8 + u];
      sRed[tid] = p;
    }
    __syncthreads();
    if (tid < MT) {
      float sc = rb;
#pragma unroll
      for (int i = 0; i < 16; ++i) sc += sRed[tid * 16 + i];
      recon[(size_t)(bn0 + tid) * Lc + t] = sc;
    }
    if (t == Lc - 1) {   // uniform branch
      __syncthreads();
      int m = tid >> 4, sub = tid & 15;
      float p = 0.f;
#pragma unroll
      for (int u = 0; u < 8; ++u) p += sHf[m * Hc + sub * 8 + u] * sPw[sub * 8 + u];
      sRed[tid] = p;
      __syncthreads();
      if (tid < MT) {
        float sc = pbv;
#pragma unroll
        for (int i = 0; i < 16; ++i) sc += sRed[tid * 16 + i];
        pred[bn0 + tid] = sc;
      }
    }
    __syncthreads();
  }
}

// =================================================================
extern "C" void kernel_launch(void* const* d_in, const int* in_sizes, int n_in,
                              void* d_out, int out_size, void* d_ws, size_t ws_size,
                              hipStream_t stream) {
  (void)in_sizes; (void)n_in; (void)out_size; (void)ws_size;
  const float* x        = (const float*)d_in[0];
  const float* enc_Wih  = (const float*)d_in[1];
  const float* enc_Whh  = (const float*)d_in[2];
  const float* enc_bih  = (const float*)d_in[3];
  const float* enc_bhh  = (const float*)d_in[4];
  const float* pool_w   = (const float*)d_in[5];
  const float* pool_b   = (const float*)d_in[6];
  const float* Wq       = (const float*)d_in[7];
  const float* bq       = (const float*)d_in[8];
  const float* Wk       = (const float*)d_in[9];
  const float* bk       = (const float*)d_in[10];
  const float* Wv       = (const float*)d_in[11];
  const float* bv       = (const float*)d_in[12];
  const float* dec_Wih  = (const float*)d_in[13];
  const float* dec_Whh  = (const float*)d_in[14];
  const float* dec_bih  = (const float*)d_in[15];
  const float* dec_bhh  = (const float*)d_in[16];
  const float* rec_w    = (const float*)d_in[17];
  const float* rec_b    = (const float*)d_in[18];
  const float* pred_w   = (const float*)d_in[19];
  const float* pred_b   = (const float*)d_in[20];

  // ---- workspace carve ----
  char* ws = (char*)d_ws;
  bf16_t* enc_whh_bf = (bf16_t*)ws;            // 65536
  bf16_t* dec_whh_bf = enc_whh_bf + 65536;     // 65536
  bf16_t* dec_wih_bf = dec_whh_bf + 65536;     // 65536
  bf16_t* wq_bf      = dec_wih_bf + 65536;     // 16384
  bf16_t* wk_bf      = wq_bf + 16384;          // 16384
  bf16_t* wv_bf      = wk_bf + 16384;          // 16384
  float*  c_pool     = (float*)(ws + 524288);  // [BN,128] fp32

  // ---- output carve (recon, pred, c_star, A concatenated) ----
  float* out   = (float*)d_out;
  float* recon = out;                      // 32*64*128 = 262144
  float* pred  = out + 262144;             // 2048
  float* cstar = out + 264192;             // 262144
  float* Aout  = out + 526336;             // 131072

  cvt_bf16_kernel<<<(65536 + 255) / 256, 256, 0, stream>>>(enc_Whh, enc_whh_bf, 65536);
  cvt_bf16_kernel<<<(65536 + 255) / 256, 256, 0, stream>>>(dec_Whh, dec_whh_bf, 65536);
  cvt_bf16_kernel<<<(65536 + 255) / 256, 256, 0, stream>>>(dec_Wih, dec_wih_bf, 65536);
  cvt_bf16_kernel<<<(16384 + 255) / 256, 256, 0, stream>>>(Wq, wq_bf, 16384);
  cvt_bf16_kernel<<<(16384 + 255) / 256, 256, 0, stream>>>(Wk, wk_bf, 16384);
  cvt_bf16_kernel<<<(16384 + 255) / 256, 256, 0, stream>>>(Wv, wv_bf, 16384);

  enc_lstm_kernel<<<BNc / MT, 256, ENC_LDS, stream>>>(
      x, enc_Wih, enc_bih, enc_bhh, pool_w, pool_b, enc_whh_bf, c_pool);

  attn_kernel<<<Bc, 256, ATT_LDS, stream>>>(
      c_pool, wq_bf, bq, wk_bf, bk, wv_bf, bv, cstar, Aout);

  dec_lstm_kernel<<<BNc / MT, 256, DEC_LDS, stream>>>(
      cstar, dec_wih_bf, dec_whh_bf, dec_bih, dec_bhh,
      rec_w, rec_b, pred_w, pred_b, recon, pred);
}
